// Correlation3D_35648228557129
// MI455X (gfx1250) — matrix operand order repaired and checked by reading, not compiled, not run on hardware
//
#include <hip/hip_runtime.h>
#include <hip/hip_bf16.h>

typedef __attribute__((ext_vector_type(16))) _Float16 v16h;
typedef __attribute__((ext_vector_type(8)))  float    v8f;

#define BATCH 4
#define NPTS  4096
#define CCH   64
#define SNB   16

static __device__ __forceinline__ v8f wmma_f16(v16h a, v16h b, v8f c) {
  return __builtin_amdgcn_wmma_f32_16x16x32_f16(false, a, false, b, (short)0, c, false, false);
}

// A fragment: packed buffer laid out as frag-major; one contiguous 32B load per lane.
static __device__ __forceinline__ v16h load_afrag(const _Float16* buf, int mt, int kk,
                                                  int Ktiles, int lane) {
  return *(const v16h*)(buf + (((size_t)(mt * Ktiles + kk)) * 32 + lane) * 16);
}

// B fragment from LDS staged column-major: stage[col][K], per-col stride Ks halves.
static __device__ __forceinline__ v16h load_bfrag(const _Float16* stage, int Ks, int kk,
                                                  int col, int hi) {
  return *(const v16h*)(stage + (size_t)col * Ks + kk * 32 + hi * 16);
}

// ---------------------------------------------------------------------------
// Weight repack: fp32 [Mrows][Cin] row-major -> f16 A-fragment order, K padded.
// dst[fr*512 + lane*16 + j], fr = mt*Ktiles + kk,
// M = mt*16 + (lane&15), K = kk*32 + (lane&16) + j
// ---------------------------------------------------------------------------
__global__ void pack_frags(const float* __restrict__ src, _Float16* __restrict__ dst,
                           int Mrows, int Cin, int Mtiles, int Ktiles) {
  int t = blockIdx.x * 256 + threadIdx.x;
  int total = Mtiles * Ktiles * 512;
  if (t >= total) return;
  int j    = t & 15;
  int lane = (t >> 4) & 31;
  int fr   = t >> 9;
  int kk   = fr % Ktiles;
  int mt   = fr / Ktiles;
  int m = mt * 16 + (lane & 15);
  int K = kk * 32 + (lane & 16) + j;
  float v = (m < Mrows && K < Cin) ? src[m * Cin + K] : 0.f;
  dst[t] = (_Float16)v;
}

// ---------------------------------------------------------------------------
// kNN: one thread per query; candidates tiled through LDS; sorted insertion.
// ---------------------------------------------------------------------------
__global__ __launch_bounds__(256) void knn_kernel(const float* __restrict__ qxyz,
                                                  const float* __restrict__ dxyz,
                                                  int* __restrict__ out_idx) {
  __shared__ float sx[256], sy[256], sz[256];
  const int bpb = NPTS / 256;                 // blocks per batch
  const int b = blockIdx.x / bpb;
  const int n = (blockIdx.x % bpb) * 256 + threadIdx.x;

  const float qx = qxyz[(b * 3 + 0) * NPTS + n];
  const float qy = qxyz[(b * 3 + 1) * NPTS + n];
  const float qz = qxyz[(b * 3 + 2) * NPTS + n];

  float bd[SNB];
  int   bi[SNB];
#pragma unroll
  for (int t = 0; t < SNB; ++t) { bd[t] = 3.0e38f; bi[t] = 0; }

  for (int tile = 0; tile < NPTS; tile += 256) {
    __syncthreads();
    sx[threadIdx.x] = dxyz[(b * 3 + 0) * NPTS + tile + threadIdx.x];
    sy[threadIdx.x] = dxyz[(b * 3 + 1) * NPTS + tile + threadIdx.x];
    sz[threadIdx.x] = dxyz[(b * 3 + 2) * NPTS + tile + threadIdx.x];
    __syncthreads();
    for (int j = 0; j < 256; ++j) {
      float dx = qx - sx[j], dy = qy - sy[j], dz = qz - sz[j];
      float d2 = dx * dx + dy * dy + dz * dz;
      if (d2 < bd[SNB - 1]) {
        int pos = 0;
#pragma unroll
        for (int t = 0; t < SNB; ++t) pos += (d2 >= bd[t]) ? 1 : 0;
#pragma unroll
        for (int t = SNB - 1; t > 0; --t) {
          bool mv = (t > pos);
          bd[t] = mv ? bd[t - 1] : bd[t];
          bi[t] = mv ? bi[t - 1] : bi[t];
        }
#pragma unroll
        for (int t = 0; t < SNB; ++t)
          if (t == pos) { bd[t] = d2; bi[t] = tile + j; }
      }
    }
  }
#pragma unroll
  for (int t = 0; t < SNB; ++t) out_idx[((size_t)b * NPTS + n) * SNB + t] = bi[t];
}

// ---------------------------------------------------------------------------
// group1: one wave = one query point. 42 WMMAs per wave.
// ---------------------------------------------------------------------------
struct __align__(32) ScratchG1 {
  _Float16 vin[16][160];   // v_in: [pts(64); nbr(64); diff(3); pad] col-major
  _Float16 kst[16][64];    // (v + pos_enc) f16
  _Float16 hst[16][64];    // (q - k) f16
  _Float16 posst[16][32];  // pos (5 ch, padded to 32)
  float    vst[16][64];    // v fp32, for attn-weighted sum
  float    red[32];
  float    attnv[16];
  float    attw[16];
  int      sidx[16];
};

__global__ __launch_bounds__(128) void group1_kernel(
    const float* __restrict__ xyz1, const float* __restrict__ feat1,
    const float* __restrict__ xyz2, const float* __restrict__ feat2,
    const _Float16* __restrict__ fq, const _Float16* __restrict__ fk,
    const _Float16* __restrict__ fv, const _Float16* __restrict__ fpos,
    const _Float16* __restrict__ fmlp1, const float* __restrict__ w_mlp2,
    const float* __restrict__ b_pos1, const int* __restrict__ idx,
    float* __restrict__ p2n) {
  __shared__ ScratchG1 ws[4];
  const int wave = threadIdx.x >> 5;
  const int lane = threadIdx.x & 31;
  const int col  = lane & 15;
  const int hi   = lane >> 4;
  ScratchG1& sc = ws[wave];

  const int gq = blockIdx.x * 4 + wave;
  const int b  = gq / NPTS;
  const int n  = gq % NPTS;

  if (lane < 16) sc.sidx[col] = idx[((size_t)b * NPTS + n) * SNB + col];
  __syncthreads();

  // rows 0..63: query features broadcast to all 16 columns
  {
    float pf0 = feat1[(b * CCH + lane * 2 + 0) * NPTS + n];
    float pf1 = feat1[(b * CCH + lane * 2 + 1) * NPTS + n];
    _Float16 h0 = (_Float16)pf0, h1 = (_Float16)pf1;
#pragma unroll
    for (int s = 0; s < 16; ++s) {
      sc.vin[s][lane * 2 + 0] = h0;
      sc.vin[s][lane * 2 + 1] = h1;
    }
  }
  // rows 64..127: gathered neighbor features + feat_norm partials
  {
    int s = lane >> 1, ch = (lane & 1) * 32;
    int m = sc.sidx[s];
    float acc = 0.f;
    for (int c = 0; c < 32; ++c) {
      float f = feat2[(b * CCH + ch + c) * NPTS + m];
      float p = feat1[(b * CCH + ch + c) * NPTS + n];
      float d = p - f;
      acc += d * d;
      sc.vin[s][64 + ch + c] = (_Float16)f;
    }
    sc.red[lane] = acc;
  }
  __syncthreads();
  // rows 128..130: xyz diff; 131..159 zero; pos staging (fn, xn, diff, pad)
  if (lane < 16) {
    int m = sc.sidx[col];
    float dx = xyz1[(b * 3 + 0) * NPTS + n] - xyz2[(b * 3 + 0) * NPTS + m];
    float dy = xyz1[(b * 3 + 1) * NPTS + n] - xyz2[(b * 3 + 1) * NPTS + m];
    float dz = xyz1[(b * 3 + 2) * NPTS + n] - xyz2[(b * 3 + 2) * NPTS + m];
    float xn = sqrtf(dx * dx + dy * dy + dz * dz);
    float fn = sqrtf(sc.red[col * 2] + sc.red[col * 2 + 1]);
    sc.vin[col][128] = (_Float16)dx;
    sc.vin[col][129] = (_Float16)dy;
    sc.vin[col][130] = (_Float16)dz;
    for (int r = 131; r < 160; ++r) sc.vin[col][r] = (_Float16)0.f;
    sc.posst[col][0] = (_Float16)fn;
    sc.posst[col][1] = (_Float16)xn;
    sc.posst[col][2] = (_Float16)dx;
    sc.posst[col][3] = (_Float16)dy;
    sc.posst[col][4] = (_Float16)dz;
    for (int r = 5; r < 32; ++r) sc.posst[col][r] = (_Float16)0.f;
  }
  __syncthreads();

  // v = W_v @ v_in (K=160, 5 steps); pos_enc = W_pos @ pos + b
  for (int mt = 0; mt < 4; ++mt) {
    v8f z = {};
    v8f pe = wmma_f16(load_afrag(fpos, mt, 0, 1, lane),
                      load_bfrag(&sc.posst[0][0], 32, 0, col, hi), z);
    v8f va = {};
#pragma unroll
    for (int kk = 0; kk < 5; ++kk)
      va = wmma_f16(load_afrag(fv, mt, kk, 5, lane),
                    load_bfrag(&sc.vin[0][0], 160, kk, col, hi), va);
#pragma unroll
    for (int r = 0; r < 8; ++r) {
      int ch = mt * 16 + hi * 8 + r;
      sc.vst[col][ch] = va[r];
      sc.kst[col][ch] = (_Float16)(va[r] + pe[r] + b_pos1[ch]);
    }
  }
  __syncthreads();

  // q = W_q @ pts_rep ; k = W_k @ (v+pos_enc) ; stage (q-k)
  for (int mt = 0; mt < 4; ++mt) {
    v8f qa = {}, ka = {};
#pragma unroll
    for (int kk = 0; kk < 2; ++kk) {
      qa = wmma_f16(load_afrag(fq, mt, kk, 2, lane),
                    load_bfrag(&sc.vin[0][0], 160, kk, col, hi), qa);
      ka = wmma_f16(load_afrag(fk, mt, kk, 2, lane),
                    load_bfrag(&sc.kst[0][0], 64, kk, col, hi), ka);
    }
#pragma unroll
    for (int r = 0; r < 8; ++r)
      sc.hst[col][mt * 16 + hi * 8 + r] = (_Float16)(qa[r] - ka[r]);
  }
  __syncthreads();

  // mlp1 (relu) + mlp2 dot -> attention logit
  {
    v8f ha = {};
#pragma unroll
    for (int kk = 0; kk < 2; ++kk)
      ha = wmma_f16(load_afrag(fmlp1, 0, kk, 2, lane),
                    load_bfrag(&sc.hst[0][0], 64, kk, col, hi), ha);
    float part = 0.f;
#pragma unroll
    for (int r = 0; r < 8; ++r)
      part += fmaxf(ha[r], 0.f) * w_mlp2[hi * 8 + r];
    sc.red[lane] = part;
  }
  __syncthreads();
  if (lane < 16) sc.attnv[col] = sc.red[col] + sc.red[col + 16];
  __syncthreads();
  {
    float mx = -3.0e38f;
#pragma unroll
    for (int s = 0; s < 16; ++s) mx = fmaxf(mx, sc.attnv[s]);
    float sum = 0.f;
#pragma unroll
    for (int s = 0; s < 16; ++s) sum += __expf(sc.attnv[s] - mx);
    if (lane < 16) sc.attw[col] = __expf(sc.attnv[col] - mx) / sum;
  }
  __syncthreads();
  // out[c] = sum_s attn[s]*v[c,s]; leaky(0.1)
#pragma unroll
  for (int cc = 0; cc < 2; ++cc) {
    int ch = lane * 2 + cc;
    float acc = 0.f;
#pragma unroll
    for (int s = 0; s < 16; ++s) acc += sc.attw[s] * sc.vst[s][ch];
    acc = (acc > 0.f) ? acc : 0.1f * acc;
    p2n[((size_t)b * CCH + ch) * NPTS + n] = acc;
  }
}

// ---------------------------------------------------------------------------
// group2: one wave = one query point. 20 WMMAs per wave.
// ---------------------------------------------------------------------------
struct __align__(32) ScratchG2 {
  _Float16 nst[16][64];    // gathered neighbor p2n, f16, col-major
  _Float16 kst[16][64];    // nbr + pos_enc
  _Float16 posst[16][32];
  float    vst[16][64];
  float    pts[64];        // p2n column n
  float    qv[64];         // q vector
  float    red[32];
  float    attnv[16];
  float    attw[16];
  int      sidx[16];
};

__global__ __launch_bounds__(128) void group2_kernel(
    const float* __restrict__ xyz1, const float* __restrict__ p2n,
    const _Float16* __restrict__ fk, const _Float16* __restrict__ fv,
    const _Float16* __restrict__ fpos, const float* __restrict__ w_q2,
    const float* __restrict__ b_pos2, const int* __restrict__ idx,
    float* __restrict__ out) {
  __shared__ ScratchG2 ws[4];
  const int wave = threadIdx.x >> 5;
  const int lane = threadIdx.x & 31;
  const int col  = lane & 15;
  const int hi   = lane >> 4;
  ScratchG2& sc = ws[wave];

  const int gq = blockIdx.x * 4 + wave;
  const int b  = gq / NPTS;
  const int n  = gq % NPTS;

  if (lane < 16) sc.sidx[col] = idx[((size_t)b * NPTS + n) * SNB + col];
  sc.pts[lane * 2 + 0] = p2n[((size_t)b * CCH + lane * 2 + 0) * NPTS + n];
  sc.pts[lane * 2 + 1] = p2n[((size_t)b * CCH + lane * 2 + 1) * NPTS + n];
  __syncthreads();

  // gather neighbor features
  {
    int s = lane >> 1, ch = (lane & 1) * 32;
    int m = sc.sidx[s];
    for (int c = 0; c < 32; ++c)
      sc.nst[s][ch + c] = (_Float16)p2n[((size_t)b * CCH + ch + c) * NPTS + m];
  }
  // pos = [diff(3); norm] padded to 32
  if (lane < 16) {
    int m = sc.sidx[col];
    float dx = xyz1[(b * 3 + 0) * NPTS + n] - xyz1[(b * 3 + 0) * NPTS + m];
    float dy = xyz1[(b * 3 + 1) * NPTS + n] - xyz1[(b * 3 + 1) * NPTS + m];
    float dz = xyz1[(b * 3 + 2) * NPTS + n] - xyz1[(b * 3 + 2) * NPTS + m];
    float xn = sqrtf(dx * dx + dy * dy + dz * dz);
    sc.posst[col][0] = (_Float16)dx;
    sc.posst[col][1] = (_Float16)dy;
    sc.posst[col][2] = (_Float16)dz;
    sc.posst[col][3] = (_Float16)xn;
    for (int r = 4; r < 32; ++r) sc.posst[col][r] = (_Float16)0.f;
  }
  __syncthreads();

  // q = W_q2 @ pts (matrix-vector, VALU)
#pragma unroll
  for (int cc = 0; cc < 2; ++cc) {
    int ch = lane * 2 + cc;
    float acc = 0.f;
    for (int ci = 0; ci < 64; ++ci) acc += w_q2[ch * 64 + ci] * sc.pts[ci];
    sc.qv[ch] = acc;
  }
  // v = W_v2 @ nbr ; pos_enc ; kin staging
  for (int mt = 0; mt < 4; ++mt) {
    v8f z = {};
    v8f pe = wmma_f16(load_afrag(fpos, mt, 0, 1, lane),
                      load_bfrag(&sc.posst[0][0], 32, 0, col, hi), z);
    v8f va = {};
#pragma unroll
    for (int kk = 0; kk < 2; ++kk)
      va = wmma_f16(load_afrag(fv, mt, kk, 2, lane),
                    load_bfrag(&sc.nst[0][0], 64, kk, col, hi), va);
#pragma unroll
    for (int r = 0; r < 8; ++r) {
      int ch = mt * 16 + hi * 8 + r;
      sc.vst[col][ch] = va[r];
      sc.kst[col][ch] = (_Float16)((float)sc.nst[col][ch] + pe[r] + b_pos2[ch]);
    }
  }
  __syncthreads();

  // k = W_k2 @ kin ; attn partial = sum_c q[c]*k[c,s]
  {
    float part = 0.f;
    for (int mt = 0; mt < 4; ++mt) {
      v8f ka = {};
#pragma unroll
      for (int kk = 0; kk < 2; ++kk)
        ka = wmma_f16(load_afrag(fk, mt, kk, 2, lane),
                      load_bfrag(&sc.kst[0][0], 64, kk, col, hi), ka);
#pragma unroll
      for (int r = 0; r < 8; ++r)
        part += sc.qv[mt * 16 + hi * 8 + r] * ka[r];
    }
    sc.red[lane] = part;
  }
  __syncthreads();
  if (lane < 16) sc.attnv[col] = (sc.red[col] + sc.red[col + 16]) * 0.125f; // 64^-0.5
  __syncthreads();
  {
    float mx = -3.0e38f;
#pragma unroll
    for (int s = 0; s < 16; ++s) mx = fmaxf(mx, sc.attnv[s]);
    float sum = 0.f;
#pragma unroll
    for (int s = 0; s < 16; ++s) sum += __expf(sc.attnv[s] - mx);
    if (lane < 16) sc.attw[col] = __expf(sc.attnv[col] - mx) / sum;
  }
  __syncthreads();
#pragma unroll
  for (int cc = 0; cc < 2; ++cc) {
    int ch = lane * 2 + cc;
    float acc = 0.f;
#pragma unroll
    for (int s = 0; s < 16; ++s) acc += sc.attw[s] * sc.vst[s][ch];
    acc = (acc > 0.f) ? acc : 0.1f * acc;
    out[((size_t)b * CCH + ch) * NPTS + n] = acc;
  }
}

// ---------------------------------------------------------------------------
extern "C" void kernel_launch(void* const* d_in, const int* in_sizes, int n_in,
                              void* d_out, int out_size, void* d_ws, size_t ws_size,
                              hipStream_t stream) {
  const float* xyz1   = (const float*)d_in[0];
  const float* feat1  = (const float*)d_in[1];
  const float* xyz2   = (const float*)d_in[2];
  const float* feat2  = (const float*)d_in[3];
  const float* w_q1   = (const float*)d_in[4];
  const float* w_k1   = (const float*)d_in[5];
  const float* w_v1   = (const float*)d_in[6];
  const float* w_mlp1 = (const float*)d_in[7];
  const float* w_mlp2 = (const float*)d_in[8];
  const float* w_pos1 = (const float*)d_in[9];
  const float* b_pos1 = (const float*)d_in[10];
  const float* w_q2   = (const float*)d_in[11];
  const float* w_k2   = (const float*)d_in[12];
  const float* w_v2   = (const float*)d_in[13];
  const float* w_pos2 = (const float*)d_in[14];
  const float* b_pos2 = (const float*)d_in[15];

  char* w = (char*)d_ws;
  int* idx12 = (int*)w;      w += (size_t)BATCH * NPTS * SNB * sizeof(int);
  int* idx11 = (int*)w;      w += (size_t)BATCH * NPTS * SNB * sizeof(int);
  float* p2n = (float*)w;    w += (size_t)BATCH * CCH * NPTS * sizeof(float);
  _Float16* fq1   = (_Float16*)w; w += 4 * 2 * 512 * sizeof(_Float16);
  _Float16* fk1   = (_Float16*)w; w += 4 * 2 * 512 * sizeof(_Float16);
  _Float16* fv1   = (_Float16*)w; w += 4 * 5 * 512 * sizeof(_Float16);
  _Float16* fpos1 = (_Float16*)w; w += 4 * 1 * 512 * sizeof(_Float16);
  _Float16* fmlp1 = (_Float16*)w; w += 1 * 2 * 512 * sizeof(_Float16);
  _Float16* fk2   = (_Float16*)w; w += 4 * 2 * 512 * sizeof(_Float16);
  _Float16* fv2   = (_Float16*)w; w += 4 * 2 * 512 * sizeof(_Float16);
  _Float16* fpos2 = (_Float16*)w; w += 4 * 1 * 512 * sizeof(_Float16);

  auto pack = [&](const float* src, _Float16* dst, int Mrows, int Cin,
                  int Mtiles, int Ktiles) {
    int total = Mtiles * Ktiles * 512;
    pack_frags<<<(total + 255) / 256, 256, 0, stream>>>(src, dst, Mrows, Cin,
                                                        Mtiles, Ktiles);
  };
  pack(w_q1,   fq1,   64, 64,  4, 2);
  pack(w_k1,   fk1,   64, 64,  4, 2);
  pack(w_v1,   fv1,   64, 131, 4, 5);   // K padded 131 -> 160
  pack(w_pos1, fpos1, 64, 5,   4, 1);   // K padded 5 -> 32
  pack(w_mlp1, fmlp1, 16, 64,  1, 2);
  pack(w_k2,   fk2,   64, 64,  4, 2);
  pack(w_v2,   fv2,   64, 64,  4, 2);
  pack(w_pos2, fpos2, 64, 4,   4, 1);   // K padded 4 -> 32

  dim3 knn_grid(BATCH * (NPTS / 256));
  knn_kernel<<<knn_grid, 256, 0, stream>>>(xyz1, xyz2, idx12);
  knn_kernel<<<knn_grid, 256, 0, stream>>>(xyz1, xyz1, idx11);

  dim3 grp_grid(BATCH * NPTS / 4);
  group1_kernel<<<grp_grid, 128, 0, stream>>>(xyz1, feat1, xyz2, feat2, fq1, fk1,
                                              fv1, fpos1, fmlp1, w_mlp2, b_pos1,
                                              idx12, p2n);
  group2_kernel<<<grp_grid, 128, 0, stream>>>(xyz1, p2n, fk2, fv2, fpos2, w_q2,
                                              b_pos2, idx11, (float*)d_out);
}
